// Transformer_34574486732910
// MI455X (gfx1250) — matrix operand order repaired and checked by reading, not compiled, not run on hardware
//
#include <hip/hip_runtime.h>
#include <math.h>

typedef __attribute__((ext_vector_type(16))) _Float16 v16h;
typedef __attribute__((ext_vector_type(8)))  _Float16 v8h;
typedef __attribute__((ext_vector_type(8)))  float    v8f;

union frag16 {
    v16h v;
    v8h  h[2];
};

__device__ inline v8f wmma_f32_f16(const frag16& a, const frag16& b, v8f c) {
    // D = A(16x32 f16) * B(32x16 f16) + C(16x16 f32)
    return __builtin_amdgcn_wmma_f32_16x16x32_f16(
        /*neg_a=*/false, a.v, /*neg_b=*/false, b.v,
        /*c_mod=*/(short)0, c, /*reuse_a=*/false, /*reuse_b=*/false);
}

#define N_TOK   1024   // H*W
#define HIDDEN_ 512
#define HD      64
#define CHUNK   128

// Tensor Data Mover availability (device pass only; arity differs by toolchain)
#if defined(__HIP_DEVICE_COMPILE__) && defined(__gfx1250__)
#if __has_builtin(__builtin_amdgcn_tensor_load_to_lds) && __has_builtin(__builtin_amdgcn_s_wait_tensorcnt)
#define USE_TDM 1
#endif
#endif

#ifdef USE_TDM
typedef __attribute__((ext_vector_type(4))) unsigned int u32x4;
typedef __attribute__((ext_vector_type(8))) int          i32x8;
typedef __attribute__((ext_vector_type(4))) int          i32x4;
#endif

// ---------------------------------------------------------------------------
// Projection: out[m, n] = f16( sum_c x[b, c, m%1024] * w[c, n] + bias[n] )
// x: [B, 512, 1024] f32, w: [512, 512] f32, out: [8192, 512] f16
// Block: 256 threads (8 waves), tile 128(M) x 128(N), K-step 32.
// ---------------------------------------------------------------------------
__global__ __launch_bounds__(256) void proj_wmma_kernel(
    const float* __restrict__ x, const float* __restrict__ w,
    const float* __restrict__ bias, _Float16* __restrict__ outh)
{
    __shared__ _Float16 Ash[128][40];   // [m_local][k_local], 80B rows
    __shared__ _Float16 WTsh[128][40];  // [n_local][k_local]

    const int tid  = threadIdx.x;
    const int wave = tid >> 5;
    const int lane = tid & 31;
    const int l16  = lane & 15;
    const int half = lane >> 4;

    const int m0     = blockIdx.x * 128;      // global token-row base
    const int n0     = blockIdx.y * 128;      // output feature base
    const int b      = m0 >> 10;              // batch (128 | 1024)
    const int n_in_b = m0 & (N_TOK - 1);

    v8f acc[8];
    v8f zero = {};
#pragma unroll
    for (int t = 0; t < 8; ++t) acc[t] = zero;

    for (int k0 = 0; k0 < HIDDEN_; k0 += 32) {
        __syncthreads();
        // Stage A (transposed from [c][n]) and W (transposed to [n][k]) as f16
        for (int i = tid; i < 32 * 128; i += 256) {
            const int kk = i >> 7;        // 0..31
            const int nl = i & 127;       // 0..127
            Ash[nl][kk]  = (_Float16)x[((size_t)b * HIDDEN_ + k0 + kk) * N_TOK + n_in_b + nl];
            WTsh[nl][kk] = (_Float16)w[(size_t)(k0 + kk) * HIDDEN_ + n0 + nl];
        }
        // Prefetch next K-slab into WGP-near caches (locality 3 -> WGP scope:
        // DEV/SYS-scope prefetch would skip the WGP$, but *this* WGP consumes it)
        if (k0 + 32 < HIDDEN_) {
            const int kk = tid & 31, sg = tid >> 5;
            if (sg < 4)
                __builtin_prefetch(&x[((size_t)b * HIDDEN_ + k0 + 32 + kk) * N_TOK + n_in_b + sg * 32], 0, 3);
            else
                __builtin_prefetch(&w[(size_t)(k0 + 32 + kk) * HIDDEN_ + n0 + (sg - 4) * 32], 0, 3);
        }
        __syncthreads();

        // A fragment: lane row = wave*16 + l16; elems 0..15 -> K=half*16+e
        frag16 af;
        const _Float16* ap = &Ash[wave * 16 + l16][0];
        af.h[0] = *(const v8h*)(ap + half * 16);
        af.h[1] = *(const v8h*)(ap + half * 16 + 8);

        // Load ALL B fragments first so DS loads stay in flight, then issue WMMAs.
        frag16 bf[8];
#pragma unroll
        for (int t = 0; t < 8; ++t) {
            const _Float16* bp = &WTsh[t * 16 + l16][half * 16];
            bf[t].h[0] = *(const v8h*)(bp);
            bf[t].h[1] = *(const v8h*)(bp + 8);
        }
#pragma unroll
        for (int t = 0; t < 8; ++t) acc[t] = wmma_f32_f16(af, bf[t], acc[t]);
    }

    // Store with bias. C layout: lane holds col l16, rows j + 8*half.
    const int mrow0 = m0 + wave * 16 + 8 * half;
#pragma unroll
    for (int t = 0; t < 8; ++t) {
        const int col = n0 + t * 16 + l16;
        const float bcol = bias[col];
#pragma unroll
        for (int j = 0; j < 8; ++j) {
            outh[(size_t)(mrow0 + j) * HIDDEN_ + col] = (_Float16)(acc[t][j] + bcol);
        }
    }
}

// ---------------------------------------------------------------------------
// Flash attention per (b, head): Q,K,V f16 [8192, 512] (head slice = 64 cols).
// Block: 256 threads = 8 waves, 16 query rows per wave (128 rows/block).
// grid = (8 q-blocks, 64 b*h). Streams 128-key chunks of K/V through LDS.
// K chunk is DMA'd by the Tensor Data Mover (with LDS pad matching the bank-
// padded Ksh layout) when the TDM builtin is available.
// out: [B, 512, 1024] f32 written directly in [B,C,H,W] layout.
// ---------------------------------------------------------------------------
__global__ __launch_bounds__(256) void attn_wmma_kernel(
    const _Float16* __restrict__ Qh, const _Float16* __restrict__ Kh,
    const _Float16* __restrict__ Vh, float* __restrict__ out)
{
    __shared__ _Float16 Ksh[CHUNK][72];     // [key][d], 144B rows (64 f16 + 8 pad)
    __shared__ _Float16 VTsh[HD][136];      // [d][key], 272B rows
    __shared__ _Float16 Pst[8][16][40];     // per-wave P staging, 80B rows

    const int tid  = threadIdx.x;
    const int wave = tid >> 5;
    const int lane = tid & 31;
    const int l16  = lane & 15;
    const int half = lane >> 4;

    const int bh   = blockIdx.y;
    const int b    = bh >> 3;
    const int head = bh & 7;
    const int qrow0 = blockIdx.x * 128 + wave * 16;   // token row base for wave

    // Load Q fragments once (A layout), 16x64 split into two 16x32 fragments.
    const _Float16* qbase =
        Qh + (size_t)(b * N_TOK + qrow0 + l16) * HIDDEN_ + head * HD;
    frag16 qa0, qa1;
    qa0.h[0] = *(const v8h*)(qbase + half * 16);
    qa0.h[1] = *(const v8h*)(qbase + half * 16 + 8);
    qa1.h[0] = *(const v8h*)(qbase + 32 + half * 16);
    qa1.h[1] = *(const v8h*)(qbase + 40 + half * 16);

    v8f acc[4];
    v8f zero = {};
#pragma unroll
    for (int t = 0; t < 4; ++t) acc[t] = zero;
    float m[8], lsum[8];
#pragma unroll
    for (int j = 0; j < 8; ++j) { m[j] = -INFINITY; lsum[j] = 0.0f; }

    for (int kc = 0; kc < N_TOK; kc += CHUNK) {
        __syncthreads();   // previous chunk fully consumed before overwriting LDS

#ifdef USE_TDM
        // --- K chunk via Tensor Data Mover: 128x64 f16 tile, 8-f16 LDS row pad ---
        if (wave == 0) {
            const unsigned lds_off = (unsigned)(unsigned long long)(const void*)&Ksh[0][0];
            const unsigned long long ga =
                (unsigned long long)(const void*)(Kh + (size_t)(b * N_TOK + kc) * HIDDEN_ + head * HD);
            u32x4 g0;
            g0[0] = 1u;                                        // count=1 (valid descriptor)
            g0[1] = lds_off;                                   // lds_addr
            g0[2] = (unsigned)(ga & 0xffffffffu);              // global_addr[31:0]
            g0[3] = (unsigned)((ga >> 32) & 0x1ffffffu) | (2u << 30);  // addr[56:32] | type=2
            i32x8 g1;
            g1[0] = (1 << 16)     // data_size = 2 bytes
                  | (1 << 20)     // pad_enable
                  | (4 << 22)     // pad_interval: 32 DWORDs (=64 f16 row)
                  | (3 << 25);    // pad_amount: 4 DWORDs (=8 f16) -> 72 f16 pitch
            g1[1] = (int)(((unsigned)HIDDEN_ & 0xffffu) << 16);   // tensor_dim0[15:0]
            g1[2] = (int)(((unsigned)N_TOK & 0xffffu) << 16);     // dim0 hi=0 | tensor_dim1[15:0]
            g1[3] = (int)((unsigned)HD << 16);                    // dim1 hi=0 | tile_dim0=64
            g1[4] = CHUNK;                                        // tile_dim1=128, tile_dim2=0
            g1[5] = HIDDEN_;                                      // tensor_dim0_stride lo
            g1[6] = 0;
            g1[7] = 0;
            i32x4 gz = {0, 0, 0, 0};
#if __clang_major__ >= 23
            i32x8 z8 = {0, 0, 0, 0, 0, 0, 0, 0};
            __builtin_amdgcn_tensor_load_to_lds(g0, g1, gz, gz, z8, 0);
#else
            __builtin_amdgcn_tensor_load_to_lds(g0, g1, gz, gz, 0);
#endif
        }
        // V chunk transposed by all threads (TDM cannot transpose)
        for (int i = tid; i < CHUNK * HD; i += 256) {
            const int key = i >> 6, d = i & 63;
            VTsh[d][key] = Vh[(size_t)(b * N_TOK + kc + key) * HIDDEN_ + head * HD + d];
        }
        if (wave == 0) __builtin_amdgcn_s_wait_tensorcnt(0);
#else
        for (int i = tid; i < CHUNK * HD; i += 256) {
            const int key = i >> 6, d = i & 63;
            const size_t g = (size_t)(b * N_TOK + kc + key) * HIDDEN_ + head * HD + d;
            Ksh[key][d]  = Kh[g];
            VTsh[d][key] = Vh[g];
        }
#endif
        // Prefetch next chunk's K/V rows (one 128B line per row) into WGP-near caches
        if (kc + CHUNK < N_TOK && tid < CHUNK) {
            const size_t gp = (size_t)(b * N_TOK + kc + CHUNK + tid) * HIDDEN_ + head * HD;
            __builtin_prefetch(Kh + gp, 0, 3);
            __builtin_prefetch(Vh + gp, 0, 3);
        }
        __syncthreads();

        for (int kk = 0; kk < CHUNK; kk += 32) {
            // --- scores: two 16x16 tiles (keys kk..+15, kk+16..+31) ---
            frag16 kb00, kb01, kb10, kb11;
            {
                const _Float16* p0 = &Ksh[kk + l16][half * 16];
                kb00.h[0] = *(const v8h*)(p0);      kb00.h[1] = *(const v8h*)(p0 + 8);
                const _Float16* p1 = &Ksh[kk + l16][32 + half * 16];
                kb01.h[0] = *(const v8h*)(p1);      kb01.h[1] = *(const v8h*)(p1 + 8);
                const _Float16* p2 = &Ksh[kk + 16 + l16][half * 16];
                kb10.h[0] = *(const v8h*)(p2);      kb10.h[1] = *(const v8h*)(p2 + 8);
                const _Float16* p3 = &Ksh[kk + 16 + l16][32 + half * 16];
                kb11.h[0] = *(const v8h*)(p3);      kb11.h[1] = *(const v8h*)(p3 + 8);
            }
            v8f s0 = zero, s1 = zero;
            s0 = wmma_f32_f16(qa0, kb00, s0);
            s0 = wmma_f32_f16(qa1, kb01, s0);
            s1 = wmma_f32_f16(qa0, kb10, s1);
            s1 = wmma_f32_f16(qa1, kb11, s1);

            // V B-fragments are independent of P: issue their DS loads now so
            // they overlap the softmax VALU work below.
            frag16 vb[4];
#pragma unroll
            for (int t = 0; t < 4; ++t) {
                const _Float16* vp = &VTsh[t * 16 + l16][kk + half * 16];
                vb[t].h[0] = *(const v8h*)(vp);
                vb[t].h[1] = *(const v8h*)(vp + 8);
            }

            // --- online softmax (per row = j + 8*half; cols live across 16 lanes) ---
            float scale[8];
#pragma unroll
            for (int j = 0; j < 8; ++j) {
                float v = fmaxf(s0[j], s1[j]);
                v = fmaxf(v, __shfl_xor(v, 1, 32));
                v = fmaxf(v, __shfl_xor(v, 2, 32));
                v = fmaxf(v, __shfl_xor(v, 4, 32));
                v = fmaxf(v, __shfl_xor(v, 8, 32));
                const float mn = fmaxf(m[j], v);
                scale[j] = __expf(m[j] - mn);
                m[j] = mn;
                const float p0 = __expf(s0[j] - mn);
                const float p1 = __expf(s1[j] - mn);
                float ts = p0 + p1;
                ts += __shfl_xor(ts, 1, 32);
                ts += __shfl_xor(ts, 2, 32);
                ts += __shfl_xor(ts, 4, 32);
                ts += __shfl_xor(ts, 8, 32);
                lsum[j] = lsum[j] * scale[j] + ts;
                // Stage P as f16: row M = j + 8*half; cols l16 / 16+l16.
                Pst[wave][j + 8 * half][l16]      = (_Float16)p0;
                Pst[wave][j + 8 * half][16 + l16] = (_Float16)p1;
            }
#pragma unroll
            for (int t = 0; t < 4; ++t)
#pragma unroll
                for (int j = 0; j < 8; ++j) acc[t][j] *= scale[j];

            // Wave-internal LDS RAW: wait on DS counter before re-reading P.
            asm volatile("s_wait_dscnt 0" ::: "memory");

            // P as A fragment (16 rows x 32 keys)
            frag16 pa;
            pa.h[0] = *(const v8h*)&Pst[wave][l16][half * 16];
            pa.h[1] = *(const v8h*)&Pst[wave][l16][half * 16 + 8];

            // acc += P * V for 4 output column tiles
#pragma unroll
            for (int t = 0; t < 4; ++t) acc[t] = wmma_f32_f16(pa, vb[t], acc[t]);
        }
    }

    // Finalize: divide by lsum, store to out[b][head*64 + col][token] (contiguous in token).
    float inv[8];
#pragma unroll
    for (int j = 0; j < 8; ++j) inv[j] = 1.0f / lsum[j];
    const int ntok0 = qrow0 + 8 * half;   // rows j=0..7 are contiguous tokens
    float* op = out + ((size_t)b * HIDDEN_ + head * HD + l16) * N_TOK + ntok0;
#pragma unroll
    for (int t = 0; t < 4; ++t) {
        float4 lo = make_float4(acc[t][0] * inv[0], acc[t][1] * inv[1],
                                acc[t][2] * inv[2], acc[t][3] * inv[3]);
        float4 hi = make_float4(acc[t][4] * inv[4], acc[t][5] * inv[5],
                                acc[t][6] * inv[6], acc[t][7] * inv[7]);
        *(float4*)(op + (size_t)t * 16 * N_TOK)     = lo;
        *(float4*)(op + (size_t)t * 16 * N_TOK + 4) = hi;
    }
}

// ---------------------------------------------------------------------------
extern "C" void kernel_launch(void* const* d_in, const int* in_sizes, int n_in,
                              void* d_out, int out_size, void* d_ws, size_t ws_size,
                              hipStream_t stream) {
    const float* x  = (const float*)d_in[0];   // [8, 512, 32, 32]
    const float* wq = (const float*)d_in[1];   // [512, 512]
    const float* bq = (const float*)d_in[2];   // [512]
    const float* wk = (const float*)d_in[3];
    const float* bk = (const float*)d_in[4];
    const float* wv = (const float*)d_in[5];
    const float* bv = (const float*)d_in[6];
    float* out = (float*)d_out;                // [8, 512, 32, 32] f32

    const size_t elems = (size_t)8 * N_TOK * HIDDEN_;   // 8192 * 512
    _Float16* Qh  = (_Float16*)d_ws;
    _Float16* Khp = Qh + elems;
    _Float16* Vhp = Khp + elems;

    dim3 blk(256);
    dim3 pgrid(64, 4);   // 8192/128 M-tiles x 512/128 N-tiles
    proj_wmma_kernel<<<pgrid, blk, 0, stream>>>(x, wq, bq, Qh);
    proj_wmma_kernel<<<pgrid, blk, 0, stream>>>(x, wk, bk, Khp);
    proj_wmma_kernel<<<pgrid, blk, 0, stream>>>(x, wv, bv, Vhp);

    dim3 agrid(8, 64);   // 1024/128 q-blocks x (B*heads)
    attn_wmma_kernel<<<agrid, blk, 0, stream>>>(Qh, Khp, Vhp, out);
}
